// LayerMemoryBank_13932873908452
// MI455X (gfx1250) — compile-verified, matrix operand order checked by reading, not run on hardware
//
#include <hip/hip_runtime.h>

// ---------------------------------------------------------------------------
// LayerMemoryBank for MI455X (gfx1250): bf16 WMMA GEMM pipeline, wave32,
// double-buffered LDS staging via GLOBAL_LOAD_ASYNC_TO_LDS_B128 (ASYNCcnt).
// ---------------------------------------------------------------------------

typedef __bf16 bf16_t;
typedef __attribute__((ext_vector_type(16))) __bf16 v16bf;
typedef __attribute__((ext_vector_type(8)))  __bf16 v8bf;
typedef __attribute__((ext_vector_type(4)))  __bf16 v4bf;
typedef __attribute__((ext_vector_type(8)))  float  v8f;
typedef __attribute__((ext_vector_type(4)))  float  v4f;
typedef __attribute__((ext_vector_type(4)))  int    v4i;

#define B_    4
#define S_    4096
#define H_    2048
#define M_    64
#define NROWS (B_ * S_)        // 16384
#define GH_   1024             // H/2

#define TILE_M 128
#define TILE_N 128
#define TILE_K 64
#define LDSA_STRIDE 72         // 64 + 8 pad (bf16 elems) -> 144B rows, 16B aligned
#define LDSB_STRIDE 72

// ---------------- async global->LDS copy (16B per lane) --------------------
#if __has_builtin(__builtin_amdgcn_global_load_async_to_lds_b128)
#define HAVE_ASYNC_LDS 1
#else
#define HAVE_ASYNC_LDS 0
#endif

typedef __attribute__((address_space(1))) v4i as1_v4i;
typedef __attribute__((address_space(3))) v4i as3_v4i;

__device__ __forceinline__ void copy16_g2l(const bf16_t* __restrict__ g, bf16_t* l)
{
#if HAVE_ASYNC_LDS
    as1_v4i* gp = (as1_v4i*)(uintptr_t)(g);
    as3_v4i* lp = (as3_v4i*)(uintptr_t)(l);
    __builtin_amdgcn_global_load_async_to_lds_b128(gp, lp, 0, 0);
#else
    *(uint4*)l = *(const uint4*)g;
#endif
}

#if HAVE_ASYNC_LDS
#if __has_builtin(__builtin_amdgcn_s_wait_asynccnt)
#define WAIT_ASYNC(n) __builtin_amdgcn_s_wait_asynccnt(n)
#else
#define WAIT_ASYNC(n) asm volatile("s_wait_asynccnt %0" ::"n"(n) : "memory")
#endif
#else
#define WAIT_ASYNC(n) ((void)0)
#endif

// ---------------------------------------------------------------------------
// Generic tiled GEMM:  C[N, O] = act( A[N, K] * W^T + bias )
//   A: bf16 row-major (optionally concat of A1[..,<Ksplit] and A2[..,>=Ksplit])
//   W: bf16 [O, K] row-major; out-of-range O rows clamped (results never stored)
//   fp32 WMMA accumulate; output fp32 or bf16.
// 8 waves (4x2), each wave: 32x64 via 2x4 v_wmma_f32_16x16x32_bf16 tiles.
// Double-buffered LDS; each wave issues exactly 8 async copies per stage.
// ---------------------------------------------------------------------------
template <bool CONCAT_A, int ACT, bool OUT_BF16>
__global__ __launch_bounds__(256)
void gemm_bf16_wmma_kernel(const bf16_t* __restrict__ A1,
                           const bf16_t* __restrict__ A2,
                           int Ksplit, int lda1, int lda2,
                           const bf16_t* __restrict__ W,
                           const float* __restrict__ bias,
                           float* __restrict__ Cf,
                           bf16_t* __restrict__ Cb,
                           int K, int O)
{
    __shared__ __align__(16) bf16_t ldsA[2 * TILE_M * LDSA_STRIDE];
    __shared__ __align__(16) bf16_t ldsB[2 * TILE_N * LDSB_STRIDE];

    const int tid  = threadIdx.x;
    const int lane = tid & 31;
    const int wave = tid >> 5;
    const int wm   = wave >> 1;       // 0..3  (row block of 32)
    const int wn   = wave & 1;        // 0..1  (col block of 64)

    const int row0 = blockIdx.y * TILE_M;
    const int col0 = blockIdx.x * TILE_N;

    v8f acc[2][4];
#pragma unroll
    for (int i = 0; i < 2; ++i)
#pragma unroll
        for (int j = 0; j < 4; ++j)
            acc[i][j] = (v8f){0.f, 0.f, 0.f, 0.f, 0.f, 0.f, 0.f, 0.f};

    // ---- stage one K-tile (128x64 A + 128x64 B, bf16) into LDS buffer ----
    auto stage = [&](int buf, int k0) {
        const bf16_t* Ab = A1;
        int kk = k0, lda = lda1;
        if (CONCAT_A && k0 >= Ksplit) { Ab = A2; kk = k0 - Ksplit; lda = lda2; }
        bf16_t* la = &ldsA[buf * (TILE_M * LDSA_STRIDE)];
        bf16_t* lb = &ldsB[buf * (TILE_N * LDSB_STRIDE)];
#pragma unroll
        for (int it = 0; it < 4; ++it) {            // 1024 x 16B  (A tile)
            int idx = tid + it * 256;
            int r = idx >> 3, g = idx & 7;
            copy16_g2l(Ab + (size_t)(row0 + r) * lda + kk + 8 * g,
                       la + r * LDSA_STRIDE + 8 * g);
        }
#pragma unroll
        for (int it = 0; it < 4; ++it) {            // 1024 x 16B  (B tile)
            int idx = tid + it * 256;
            int n = idx >> 3, g = idx & 7;
            int o = col0 + n;
            if (o > O - 1) o = O - 1;               // clamp: uniform 8 ops/wave
            copy16_g2l(W + (size_t)o * K + k0 + 8 * g,
                       lb + n * LDSB_STRIDE + 8 * g);
        }
    };

    const int nstages = K / TILE_K;
    stage(0, 0);

    for (int s = 0; s < nstages; ++s) {
        const int cur = s & 1;
        const bool pref = (s + 1 < nstages);
        if (pref) stage(cur ^ 1, (s + 1) * TILE_K);
#if HAVE_ASYNC_LDS
        if (pref) { WAIT_ASYNC(8); } else { WAIT_ASYNC(0); }
#endif
        __syncthreads();

        const bf16_t* la = &ldsA[cur * (TILE_M * LDSA_STRIDE)];
        const bf16_t* lb = &ldsB[cur * (TILE_N * LDSB_STRIDE)];

#pragma unroll
        for (int ks = 0; ks < TILE_K; ks += 32) {
            const int hsel = (lane >> 4) << 3;      // lanes 16..31: K offset +8
            v16bf af[2], bf[4];
#pragma unroll
            for (int i = 0; i < 2; ++i) {
                const bf16_t* p =
                    &la[(wm * 32 + i * 16 + (lane & 15)) * LDSA_STRIDE + ks + hsel];
                v8bf lo = *(const v8bf*)p;          // K: base .. base+7
                v8bf hi = *(const v8bf*)(p + 16);   // K: base+16 .. base+23
#pragma unroll
                for (int e = 0; e < 8; ++e) { af[i][e] = lo[e]; af[i][e + 8] = hi[e]; }
            }
#pragma unroll
            for (int j = 0; j < 4; ++j) {
                const bf16_t* p =
                    &lb[(wn * 64 + j * 16 + (lane & 15)) * LDSB_STRIDE + ks + hsel];
                v8bf lo = *(const v8bf*)p;
                v8bf hi = *(const v8bf*)(p + 16);
#pragma unroll
                for (int e = 0; e < 8; ++e) { bf[j][e] = lo[e]; bf[j][e + 8] = hi[e]; }
            }
#pragma unroll
            for (int i = 0; i < 2; ++i)
#pragma unroll
                for (int j = 0; j < 4; ++j)
                    acc[i][j] = __builtin_amdgcn_wmma_f32_16x16x32_bf16(
                        false, af[i], false, bf[j], (short)0, acc[i][j], false, false);
        }
        __syncthreads();   // protects LDS buffer 'cur' before it is re-staged
    }

    // ---- epilogue: lane (0..15|16..31) -> N, VGPR t -> M (+8 hi half) ----
#pragma unroll
    for (int j = 0; j < 4; ++j) {
        int col = col0 + wn * 64 + j * 16 + (lane & 15);
        if (col >= O) continue;
        float bvv = bias ? bias[col] : 0.f;
#pragma unroll
        for (int i = 0; i < 2; ++i) {
            int rbase = row0 + wm * 32 + i * 16 + ((lane >> 4) << 3);
#pragma unroll
            for (int t = 0; t < 8; ++t) {
                float v = acc[i][j][t] + bvv;
                if (ACT == 1) v = fmaxf(v, 0.f);
                if (OUT_BF16) Cb[(size_t)(rbase + t) * O + col] = (bf16_t)v;
                else          Cf[(size_t)(rbase + t) * O + col] = v;
            }
        }
    }
}

// ---------------------------------------------------------------------------
// fp32 -> bf16 convert (4 elems/thread)
// ---------------------------------------------------------------------------
__global__ __launch_bounds__(256)
void cvt_f32_bf16_kernel(const float* __restrict__ src, bf16_t* __restrict__ dst,
                         int n4)
{
    int i = blockIdx.x * 256 + threadIdx.x;
    if (i < n4) {
        v4f d = *(const v4f*)(src + 4 * (size_t)i);
        v4bf h;
        h[0] = (bf16_t)d[0]; h[1] = (bf16_t)d[1];
        h[2] = (bf16_t)d[2]; h[3] = (bf16_t)d[3];
        *(v4bf*)(dst + 4 * (size_t)i) = h;
    }
}

// memory_values [64, 2048] fp32 -> transposed bf16 [2048, 64]
__global__ __launch_bounds__(256)
void mvT_bf16_kernel(const float* __restrict__ mv, bf16_t* __restrict__ mvT)
{
    int idx = blockIdx.x * 256 + threadIdx.x;     // 131072 elems
    int h = idx >> 6, m = idx & 63;
    mvT[idx] = (bf16_t)mv[(size_t)m * H_ + h];
}

// ---------------------------------------------------------------------------
// Softmax over the 64 memory slots: fp32 logits -> bf16 probs. Wave per row.
// ---------------------------------------------------------------------------
__global__ __launch_bounds__(256)
void softmax64_kernel(const float* __restrict__ logits, bf16_t* __restrict__ probs)
{
    int row  = blockIdx.x * 8 + (threadIdx.x >> 5);
    int lane = threadIdx.x & 31;
    const float* p = logits + (size_t)row * M_;
    float x0 = p[lane], x1 = p[lane + 32];
    float m = fmaxf(x0, x1);
    for (int off = 16; off > 0; off >>= 1) m = fmaxf(m, __shfl_xor(m, off, 32));
    float e0 = __expf(x0 - m), e1 = __expf(x1 - m);
    float s = e0 + e1;
    for (int off = 16; off > 0; off >>= 1) s += __shfl_xor(s, off, 32);
    float inv = 1.f / s;
    bf16_t* q = probs + (size_t)row * M_;
    q[lane]      = (bf16_t)(e0 * inv);
    q[lane + 32] = (bf16_t)(e1 * inv);
}

// ---------------------------------------------------------------------------
// reuse_gate = sigmoid(g_hidden . Wg2 + bg2). One wave per row (1024 dot).
// ---------------------------------------------------------------------------
__global__ __launch_bounds__(256)
void gate_kernel(const bf16_t* __restrict__ gh, const float* __restrict__ Wg2,
                 const float* __restrict__ bg2, float* __restrict__ gate,
                 float* __restrict__ out_gate)
{
    int row  = blockIdx.x * 8 + (threadIdx.x >> 5);
    int lane = threadIdx.x & 31;
    const bf16_t* r = gh + (size_t)row * GH_;
    float s = 0.f;
#pragma unroll 4
    for (int i = lane; i < GH_; i += 32) s += (float)r[i] * Wg2[i];
    for (int off = 16; off > 0; off >>= 1) s += __shfl_xor(s, off, 32);
    if (lane == 0) {
        float g = 1.f / (1.f + __expf(-(s + bg2[0])));
        gate[row]     = g;
        out_gate[row] = g;
    }
}

// ---------------------------------------------------------------------------
// fused = (1-g)*current + g*retrieved, then LayerNorm. One block per row.
// ---------------------------------------------------------------------------
__global__ __launch_bounds__(256)
void fuse_ln_kernel(const float* __restrict__ cur, const bf16_t* __restrict__ retr,
                    const float* __restrict__ gate, const float* __restrict__ gamma,
                    const float* __restrict__ beta, float* __restrict__ out)
{
    __shared__ float rs[8], rs2[8];
    int row = blockIdx.x;
    float g = gate[row];
    const float*  c = cur  + (size_t)row * H_;
    const bf16_t* r = retr + (size_t)row * H_;
    float vals[8];
    float s = 0.f, s2 = 0.f;
#pragma unroll
    for (int i = 0; i < 8; ++i) {
        int idx = threadIdx.x + i * 256;
        float f = (1.f - g) * c[idx] + g * (float)r[idx];
        vals[i] = f;
        s += f;
        s2 += f * f;
    }
    for (int off = 16; off > 0; off >>= 1) {
        s  += __shfl_xor(s,  off, 32);
        s2 += __shfl_xor(s2, off, 32);
    }
    int wv = threadIdx.x >> 5, ln = threadIdx.x & 31;
    if (ln == 0) { rs[wv] = s; rs2[wv] = s2; }
    __syncthreads();
    if (threadIdx.x == 0) {
        float S = 0.f, S2 = 0.f;
#pragma unroll
        for (int i = 0; i < 8; ++i) { S += rs[i]; S2 += rs2[i]; }
        rs[0] = S; rs2[0] = S2;
    }
    __syncthreads();
    float mu   = rs[0]  * (1.f / (float)H_);
    float var  = rs2[0] * (1.f / (float)H_) - mu * mu;
    float rinv = rsqrtf(var + 1e-5f);
#pragma unroll
    for (int i = 0; i < 8; ++i) {
        int idx = threadIdx.x + i * 256;
        out[(size_t)row * H_ + idx] = (vals[i] - mu) * rinv * gamma[idx] + beta[idx];
    }
}

// ---------------------------------------------------------------------------
// Column sums of bf16 k/v over sequence dim: fp32 partials + global atomics.
// grid: (H/256, B, 16 s-chunks of 256)
// ---------------------------------------------------------------------------
__global__ __launch_bounds__(256)
void colsum_kernel(const bf16_t* __restrict__ kq, const bf16_t* __restrict__ vq,
                   float* __restrict__ ksum, float* __restrict__ vsum)
{
    int h  = blockIdx.x * 256 + threadIdx.x;
    int b  = blockIdx.y;
    int s0 = blockIdx.z * 256;
    const bf16_t* kp = kq + ((size_t)b * S_ + s0) * H_ + h;
    const bf16_t* vp = vq + ((size_t)b * S_ + s0) * H_ + h;
    float a = 0.f, c = 0.f;
    for (int s = 0; s < 256; ++s) {
        a += (float)kp[(size_t)s * H_];
        c += (float)vp[(size_t)s * H_];
    }
    atomicAdd(&ksum[b * H_ + h], a);
    atomicAdd(&vsum[b * H_ + h], c);
}

// ---------------------------------------------------------------------------
// Copy memory bank + counters into d_out; zero column-sum accumulators.
// ---------------------------------------------------------------------------
__global__ __launch_bounds__(256)
void copy_init_kernel(const float* __restrict__ mk, const float* __restrict__ mv,
                      const int* __restrict__ uc, const int* __restrict__ ts,
                      float* __restrict__ omk, float* __restrict__ omv,
                      int* __restrict__ ouc, int* __restrict__ ots,
                      float* __restrict__ ksum, float* __restrict__ vsum)
{
    int i = blockIdx.x * 256 + threadIdx.x;
    if (i < M_ * H_) { omk[i] = mk[i]; omv[i] = mv[i]; }
    if (i < B_ * H_) { ksum[i] = 0.f; vsum[i] = 0.f; }
    if (i < M_)      { ouc[i] = uc[i]; ots[i] = ts[i]; }
}

// ---------------------------------------------------------------------------
// Sequential argmin-scatter memory update (4 steps). Single block.
// ---------------------------------------------------------------------------
__global__ __launch_bounds__(256)
void memupdate_kernel(const float* __restrict__ ksum, const float* __restrict__ vsum,
                      float* __restrict__ mk, float* __restrict__ mv,
                      int* __restrict__ uc, int* __restrict__ ts)
{
    __shared__ int s_idx, s_tmax;
    const float inv_s = 1.f / (float)S_;
    for (int b = 0; b < B_; ++b) {
        if (threadIdx.x == 0) {
            float best = 3.4e38f; int bi = 0; int tmax = -2147483647;
            for (int m = 0; m < M_; ++m) {
                float sc = (float)uc[m] + 0.1f * (float)ts[m];
                if (sc < best) { best = sc; bi = m; }      // first-min, like argmin
                tmax = (ts[m] > tmax) ? ts[m] : tmax;
            }
            s_idx = bi; s_tmax = tmax;
        }
        __syncthreads();
        int idx = s_idx;
        for (int i = threadIdx.x; i < H_; i += 256) {
            mk[(size_t)idx * H_ + i] = ksum[b * H_ + i] * inv_s;   // key mean
            mv[(size_t)idx * H_ + i] = vsum[b * H_ + i] * inv_s;   // value mean
        }
        __syncthreads();
        if (threadIdx.x == 0) { uc[idx] += 1; ts[idx] = s_tmax + 1; }
        __syncthreads();
    }
}

// ---------------------------------------------------------------------------
extern "C" void kernel_launch(void* const* d_in, const int* in_sizes, int n_in,
                              void* d_out, int out_size, void* d_ws, size_t ws_size,
                              hipStream_t stream)
{
    (void)in_sizes; (void)n_in; (void)out_size; (void)ws_size;

    const float* current  = (const float*)d_in[0];
    const float* layer_in = (const float*)d_in[1];
    const float* mem_k    = (const float*)d_in[2];
    const float* mem_v    = (const float*)d_in[3];
    const float* Wq       = (const float*)d_in[4];
    const float* bq       = (const float*)d_in[5];
    const float* Wk       = (const float*)d_in[6];
    const float* bk       = (const float*)d_in[7];
    const float* Wv       = (const float*)d_in[8];
    const float* bv       = (const float*)d_in[9];
    const float* Wg1      = (const float*)d_in[10];
    const float* bg1      = (const float*)d_in[11];
    const float* Wg2      = (const float*)d_in[12];
    const float* bg2      = (const float*)d_in[13];
    const float* ln_g     = (const float*)d_in[14];
    const float* ln_b     = (const float*)d_in[15];
    const int*   uc_in    = (const int*)d_in[16];
    const int*   ts_in    = (const int*)d_in[17];

    // ---- workspace layout (bytes) ----
    char* ws = (char*)d_ws;
    bf16_t* li_b   = (bf16_t*)(ws + 0ull);            // [16384,2048] bf16
    bf16_t* cur_b  = (bf16_t*)(ws + 67108864ull);     // [16384,2048]
    bf16_t* q_b    = (bf16_t*)(ws + 134217728ull);    // [16384,2048]; reused as gh [16384,1024]
    bf16_t* k_b    = (bf16_t*)(ws + 201326592ull);    // [16384,2048]
    bf16_t* v_b    = (bf16_t*)(ws + 268435456ull);    // [16384,2048]
    bf16_t* retr_b = (bf16_t*)(ws + 335544320ull);    // [16384,2048]
    bf16_t* Wq_b   = (bf16_t*)(ws + 402653184ull);    // [2048,2048]
    bf16_t* Wk_b   = (bf16_t*)(ws + 411041792ull);
    bf16_t* Wv_b   = (bf16_t*)(ws + 419430400ull);
    bf16_t* Wg1_b  = (bf16_t*)(ws + 427819008ull);    // [1024,4096]
    bf16_t* mk_b   = (bf16_t*)(ws + 436207616ull);    // [64,2048]
    bf16_t* mvT_b  = (bf16_t*)(ws + 436469760ull);    // [2048,64] transposed
    float*  logit_ws = (float*)(ws + 436731904ull);   // [16384,64] fp32
    bf16_t* probs_b  = (bf16_t*)(ws + 440926208ull);  // [16384,64] bf16
    float*  gate_ws  = (float*)(ws + 443023360ull);   // [16384]
    float*  ksum     = (float*)(ws + 443088896ull);   // [4,2048]
    float*  vsum     = (float*)(ws + 443121664ull);   // [4,2048]
    bf16_t* gh_b     = q_b;                           // reuse q region

    // ---- output layout (concatenated tuple, 4B elems) ----
    float* out      = (float*)d_out;                  // [4,4096,2048]
    float* out_gate = out + 33554432ull;              // [4,4096,1]
    float* out_mk   = out + 33570816ull;              // [64,2048]
    float* out_mv   = out + 33701888ull;              // [64,2048]
    int*   out_uc   = (int*)(out + 33832960ull);      // [64] int32
    int*   out_ts   = (int*)(out + 33833024ull);      // [64] int32

    dim3 blk(256);

    // ---- one-time fp32 -> bf16 packing ----
    cvt_f32_bf16_kernel<<<32768, blk, 0, stream>>>(layer_in, li_b,  NROWS * H_ / 4);
    cvt_f32_bf16_kernel<<<32768, blk, 0, stream>>>(current,  cur_b, NROWS * H_ / 4);
    cvt_f32_bf16_kernel<<<4096,  blk, 0, stream>>>(Wq,  Wq_b,  H_ * H_ / 4);
    cvt_f32_bf16_kernel<<<4096,  blk, 0, stream>>>(Wk,  Wk_b,  H_ * H_ / 4);
    cvt_f32_bf16_kernel<<<4096,  blk, 0, stream>>>(Wv,  Wv_b,  H_ * H_ / 4);
    cvt_f32_bf16_kernel<<<4096,  blk, 0, stream>>>(Wg1, Wg1_b, GH_ * 2 * H_ / 4);
    cvt_f32_bf16_kernel<<<128,   blk, 0, stream>>>(mem_k, mk_b, M_ * H_ / 4);
    mvT_bf16_kernel<<<512, blk, 0, stream>>>(mem_v, mvT_b);

    dim3 g_qkv(H_  / TILE_N, NROWS / TILE_M);         // (16,128)
    dim3 g_log(1,            NROWS / TILE_M);         // (1,128), O=64 clamped
    dim3 g_gh (GH_ / TILE_N, NROWS / TILE_M);         // (8,128)

    // q/k/v projections (async-staged bf16 WMMA, fp32 accumulate, bf16 out)
    gemm_bf16_wmma_kernel<false, 0, true><<<g_qkv, blk, 0, stream>>>(
        li_b,  nullptr, 0, H_, 0, Wq_b, bq, nullptr, q_b, H_, H_);
    gemm_bf16_wmma_kernel<false, 0, true><<<g_qkv, blk, 0, stream>>>(
        cur_b, nullptr, 0, H_, 0, Wk_b, bk, nullptr, k_b, H_, H_);
    gemm_bf16_wmma_kernel<false, 0, true><<<g_qkv, blk, 0, stream>>>(
        cur_b, nullptr, 0, H_, 0, Wv_b, bv, nullptr, v_b, H_, H_);

    // attention logits against the 64 memory keys (fp32 out for stable softmax)
    gemm_bf16_wmma_kernel<false, 0, false><<<g_log, blk, 0, stream>>>(
        q_b, nullptr, 0, H_, 0, mk_b, nullptr, logit_ws, nullptr, H_, M_);

    // softmax over the 64 slots
    softmax64_kernel<<<NROWS / 8, blk, 0, stream>>>(logit_ws, probs_b);

    // retrieved = probs @ memory_values   (W = mv^T bf16 [2048, 64])
    gemm_bf16_wmma_kernel<false, 0, true><<<g_qkv, blk, 0, stream>>>(
        probs_b, nullptr, 0, M_, 0, mvT_b, nullptr, nullptr, retr_b, M_, H_);

    // g_hidden = relu(concat(layer_input, retrieved) @ Wg1^T + bg1)
    gemm_bf16_wmma_kernel<true, 1, true><<<g_gh, blk, 0, stream>>>(
        li_b, retr_b, H_, H_, H_, Wg1_b, bg1, nullptr, gh_b, 2 * H_, GH_);

    // reuse_gate = sigmoid(g_hidden . Wg2 + bg2)
    gate_kernel<<<NROWS / 8, blk, 0, stream>>>(gh_b, Wg2, bg2, gate_ws, out_gate);

    // fused gate + LayerNorm -> main output
    fuse_ln_kernel<<<NROWS, blk, 0, stream>>>(current, retr_b, gate_ws, ln_g, ln_b, out);

    // memory bank update: copy state, column means of k/v, sequential scan
    copy_init_kernel<<<(M_ * H_ + 255) / 256, blk, 0, stream>>>(
        mem_k, mem_v, uc_in, ts_in, out_mk, out_mv, out_uc, out_ts, ksum, vsum);
    colsum_kernel<<<dim3(H_ / 256, B_, S_ / 256), blk, 0, stream>>>(k_b, v_b, ksum, vsum);
    memupdate_kernel<<<1, blk, 0, stream>>>(ksum, vsum, out_mk, out_mv, out_uc, out_ts);
}
